// Attention4D_57801669870303
// MI455X (gfx1250) — matrix operand order repaired and checked by reading, not compiled, not run on hardware
//
#include <hip/hip_runtime.h>

typedef _Float16 v16h __attribute__((ext_vector_type(16)));
typedef _Float16 v8h  __attribute__((ext_vector_type(8)));
typedef float    v8f  __attribute__((ext_vector_type(8)));

#define NPIX   3136
#define NH     8
#define CHUNK  64
#define NCHUNK 49          // 3136 / 64

static __device__ __forceinline__ v8f wmma16(v16h a, v16h b, v8f c) {
  return __builtin_amdgcn_wmma_f32_16x16x32_f16(false, a, false, b, (short)0, c,
                                                false, false);
}

// ---------------------------------------------------------------------------
// Kernel 1: fused 1x1 projections q,k,v.  q,k -> f16 [b,h,n,16] (q pre-scaled
// by dk^-0.5), v -> f32 [b,512,n] (spatial layout for the depthwise conv).
// ---------------------------------------------------------------------------
__global__ __launch_bounds__(256) void qkv_kernel(
    const float* __restrict__ x,
    const float* __restrict__ wq, const float* __restrict__ bq,
    const float* __restrict__ wk, const float* __restrict__ bk,
    const float* __restrict__ wv, const float* __restrict__ bv,
    _Float16* __restrict__ qf, _Float16* __restrict__ kf,
    float* __restrict__ vt)
{
  __shared__ float xt[128][32];
  const int tid = threadIdx.x;
  const int b   = blockIdx.x / 98;
  const int n0  = (blockIdx.x % 98) * 32;

  #pragma unroll
  for (int i = 0; i < 16; ++i) {
    int idx = tid + 256 * i;              // 4096 = 128*32
    int c = idx >> 5, j = idx & 31;
    xt[c][j] = x[((size_t)b * 128 + c) * NPIX + n0 + j];
  }
  __syncthreads();

  for (int ob = 0; ob < 3; ++ob) {
    int oc = ob * 256 + tid;              // 0..767
    const float* wrow; float bias;
    if (oc < 128)       { wrow = wq + (size_t)oc * 128;        bias = bq[oc];       }
    else if (oc < 256)  { wrow = wk + (size_t)(oc - 128) * 128; bias = bk[oc - 128]; }
    else                { wrow = wv + (size_t)(oc - 256) * 128; bias = bv[oc - 256]; }

    float acc[32];
    #pragma unroll
    for (int j = 0; j < 32; ++j) acc[j] = 0.f;
    for (int c = 0; c < 128; ++c) {
      float wvv = wrow[c];
      #pragma unroll
      for (int j = 0; j < 32; ++j) acc[j] = fmaf(wvv, xt[c][j], acc[j]);
    }

    if (oc < 128) {
      int h = oc >> 4, dk = oc & 15;
      _Float16* dst = qf + ((size_t)(b * NH + h) * NPIX + n0) * 16 + dk;
      #pragma unroll
      for (int j = 0; j < 32; ++j) dst[j * 16] = (_Float16)((acc[j] + bias) * 0.25f);
    } else if (oc < 256) {
      int o = oc - 128, h = o >> 4, dk = o & 15;
      _Float16* dst = kf + ((size_t)(b * NH + h) * NPIX + n0) * 16 + dk;
      #pragma unroll
      for (int j = 0; j < 32; ++j) dst[j * 16] = (_Float16)(acc[j] + bias);
    } else {
      int o = oc - 256;
      float* dst = vt + ((size_t)b * 512 + o) * NPIX + n0;
      #pragma unroll
      for (int j = 0; j < 32; ++j) dst[j] = acc[j] + bias;
    }
  }
}

// ---------------------------------------------------------------------------
// Kernel 2: depthwise 3x3 (SAME) + bias; output f16, layout [b][ch][n] which
// equals d-major [b][g][d][n] since ch = g*64+d.
// ---------------------------------------------------------------------------
__global__ void dw_kernel(const float* __restrict__ vt,
                          const float* __restrict__ wdw,
                          const float* __restrict__ bdw,
                          _Float16* __restrict__ vf)
{
  int idx = blockIdx.x * 256 + threadIdx.x;
  if (idx >= 2 * 512 * NPIX) return;
  int n  = idx % NPIX;
  int ch = (idx / NPIX) % 512;
  int b  = idx / (NPIX * 512);
  int hh = n / 56, ww = n % 56;
  const float* src = vt + ((size_t)b * 512 + ch) * NPIX;
  float acc = bdw[ch];
  #pragma unroll
  for (int i = 0; i < 3; ++i)
    #pragma unroll
    for (int j = 0; j < 3; ++j) {
      int y = hh + i - 1, xw = ww + j - 1;
      if (y >= 0 && y < 56 && xw >= 0 && xw < 56)
        acc = fmaf(wdw[ch * 9 + i * 3 + j], src[y * 56 + xw], acc);
    }
  vf[idx] = (_Float16)acc;
}

// ---------------------------------------------------------------------------
// Kernel 3: w_proj f32[128][512] -> f16 (same layout; used as "row=c" B source)
// ---------------------------------------------------------------------------
__global__ void cvt_kernel(const float* __restrict__ wp, _Float16* __restrict__ wpf)
{
  int i = blockIdx.x * 256 + threadIdx.x;
  if (i < 128 * 512) wpf[i] = (_Float16)wp[i];
}

// ---------------------------------------------------------------------------
// Kernel 4: fused attention. One workgroup = (b, 16 query rows); wave = head.
// Pass 1: QK^T (WMMA) -> th1 mix -> online row max/expsum.
// Pass 2: recompute QK^T -> th1 -> normalized P -> th2 mix (+b2) -> f16 ->
//         att2 @ V (WMMA, accum f32) -> ReLU -> netR f16 [b*N][512].
// ---------------------------------------------------------------------------
__global__ __launch_bounds__(256) void attn_kernel(
    const _Float16* __restrict__ qf, const _Float16* __restrict__ kf,
    const _Float16* __restrict__ vf,
    const float* __restrict__ w1g, const float* __restrict__ b1g,
    const float* __restrict__ w2g, const float* __restrict__ b2g,
    _Float16* __restrict__ netR)
{
  __shared__ float s_raw[NH * 16 * 68];                 // padded stride 68
  __shared__ __attribute__((aligned(16))) _Float16 a2s[NH * 16 * 72]; // stride 72
  __shared__ float lw1[64], lw2[64], lb1[8], lb2[8];
  __shared__ float rmax[NH * 16], rinv[NH * 16];

  const int tid  = threadIdx.x;
  const int wave = tid >> 5;          // head index
  const int lane = tid & 31;
  const int hl   = lane & 15;
  const int hi   = lane >> 4;
  const int b    = blockIdx.x / (NPIX / 16);
  const int n0   = (blockIdx.x % (NPIX / 16)) * 16;

  if (tid < 64) { lw1[tid] = w1g[tid]; lw2[tid] = w2g[tid]; }
  if (tid < 8)  { lb1[tid] = b1g[tid]; lb2[tid] = b2g[tid]; }

  // Q tile in A layout (16x32, K 16..31 zero-padded), kept in registers.
  v16h aq;
  #pragma unroll
  for (int i = 0; i < 16; ++i) aq[i] = (_Float16)0.f;
  {
    const _Float16* qrow =
        qf + ((size_t)(b * NH + wave) * NPIX + n0 + hl) * 16;
    if (hi == 0) aq = *(const v16h*)qrow;
  }
  const _Float16* kfh = kf + (size_t)(b * NH + wave) * NPIX * 16;
  const _Float16* vfh = vf + (size_t)(b * NH + wave) * 64 * NPIX;
  __syncthreads();

  float runM = -3.0e38f, runS = 0.f;

  // ------------------------- pass 1: statistics -------------------------
  for (int mc = 0; mc < NCHUNK; ++mc) {
    const int m0 = mc * CHUNK;
    #pragma unroll
    for (int sub = 0; sub < 4; ++sub) {
      v16h bk;
      #pragma unroll
      for (int i = 0; i < 16; ++i) bk[i] = (_Float16)0.f;
      if (hi == 0)
        bk = *(const v16h*)(kfh + (size_t)(m0 + sub * 16 + hl) * 16);
      v8f d = {};
      d = wmma16(aq, bk, d);
      #pragma unroll
      for (int r = 0; r < 8; ++r)
        s_raw[wave * 1088 + (r + hi * 8) * 68 + sub * 16 + hl] = d[r];
    }
    __syncthreads();
    // th1: read all heads, barrier, write mixed back in place
    float sv[8][4];
    #pragma unroll
    for (int i = 0; i < 4; ++i) {
      int s = tid + 256 * i, n = s >> 6, m = s & 63;
      #pragma unroll
      for (int h = 0; h < 8; ++h) sv[h][i] = s_raw[h * 1088 + n * 68 + m];
    }
    __syncthreads();
    #pragma unroll
    for (int i = 0; i < 4; ++i) {
      int s = tid + 256 * i, n = s >> 6, m = s & 63;
      #pragma unroll
      for (int g = 0; g < 8; ++g) {
        float acc = lb1[g];
        #pragma unroll
        for (int h = 0; h < 8; ++h) acc = fmaf(lw1[g * 8 + h], sv[h][i], acc);
        s_raw[g * 1088 + n * 68 + m] = acc;
      }
    }
    __syncthreads();
    // running softmax stats: lane owns (row=hl, half hi) of own head
    {
      float cv[32];
      #pragma unroll
      for (int j = 0; j < 32; ++j)
        cv[j] = s_raw[wave * 1088 + hl * 68 + hi * 32 + j];
      float cmax = cv[0];
      #pragma unroll
      for (int j = 1; j < 32; ++j) cmax = fmaxf(cmax, cv[j]);
      float nm = fmaxf(runM, cmax);
      float es = 0.f;
      #pragma unroll
      for (int j = 0; j < 32; ++j) es += __expf(cv[j] - nm);
      runS = runS * __expf(runM - nm) + es;
      runM = nm;
    }
    __syncthreads();
  }
  {
    float oM = __shfl_xor(runM, 16);
    float oS = __shfl_xor(runS, 16);
    float fM = fmaxf(runM, oM);
    float fS = runS * __expf(runM - fM) + oS * __expf(oM - fM);
    if (hi == 0) { rmax[wave * 16 + hl] = fM; rinv[wave * 16 + hl] = 1.0f / fS; }
  }
  __syncthreads();

  // ------------------------- pass 2: output -------------------------
  v8f zf = {};
  v8f O[4]; O[0] = zf; O[1] = zf; O[2] = zf; O[3] = zf;

  for (int mc = 0; mc < NCHUNK; ++mc) {
    const int m0 = mc * CHUNK;
    #pragma unroll
    for (int sub = 0; sub < 4; ++sub) {
      v16h bk;
      #pragma unroll
      for (int i = 0; i < 16; ++i) bk[i] = (_Float16)0.f;
      if (hi == 0)
        bk = *(const v16h*)(kfh + (size_t)(m0 + sub * 16 + hl) * 16);
      v8f d = {};
      d = wmma16(aq, bk, d);
      #pragma unroll
      for (int r = 0; r < 8; ++r)
        s_raw[wave * 1088 + (r + hi * 8) * 68 + sub * 16 + hl] = d[r];
    }
    __syncthreads();
    // th1 -> normalized P -> th2 (+b2) -> f16 att2 tile
    float sv[8][4];
    #pragma unroll
    for (int i = 0; i < 4; ++i) {
      int s = tid + 256 * i, n = s >> 6, m = s & 63;
      #pragma unroll
      for (int h = 0; h < 8; ++h) sv[h][i] = s_raw[h * 1088 + n * 68 + m];
    }
    #pragma unroll
    for (int i = 0; i < 4; ++i) {
      int s = tid + 256 * i, n = s >> 6, m = s & 63;
      float p[8];
      #pragma unroll
      for (int g = 0; g < 8; ++g) {
        float s1 = lb1[g];
        #pragma unroll
        for (int h = 0; h < 8; ++h) s1 = fmaf(lw1[g * 8 + h], sv[h][i], s1);
        p[g] = __expf(s1 - rmax[g * 16 + n]) * rinv[g * 16 + n];
      }
      #pragma unroll
      for (int gp = 0; gp < 8; ++gp) {
        float acc = lb2[gp];
        #pragma unroll
        for (int g = 0; g < 8; ++g) acc = fmaf(lw2[gp * 8 + g], p[g], acc);
        a2s[gp * 1152 + n * 72 + m] = (_Float16)acc;
      }
    }
    __syncthreads();
    // att2 @ V : K = 64 (two 32-chunks), 4 d-tiles of 16
    #pragma unroll
    for (int ks = 0; ks < 2; ++ks) {
      v16h av;
      const _Float16* arow = &a2s[wave * 1152 + hl * 72 + ks * 32 + hi * 8];
      v8h a0 = *(const v8h*)arow;
      v8h a1 = *(const v8h*)(arow + 16);
      #pragma unroll
      for (int i = 0; i < 8; ++i) { av[i] = a0[i]; av[8 + i] = a1[i]; }
      #pragma unroll
      for (int dt = 0; dt < 4; ++dt) {
        const _Float16* vrow =
            vfh + (size_t)(dt * 16 + hl) * NPIX + m0 + ks * 32 + hi * 8;
        v8h b0 = *(const v8h*)vrow;
        v8h b1 = *(const v8h*)(vrow + 16);
        v16h bv;
        #pragma unroll
        for (int i = 0; i < 8; ++i) { bv[i] = b0[i]; bv[8 + i] = b1[i]; }
        O[dt] = wmma16(av, bv, O[dt]);
      }
    }
    __syncthreads();
  }

  // ReLU + store netR [b*N][512] f16 (col = head*64 + d)
  const size_t rowbase = (size_t)b * NPIX + n0;
  #pragma unroll
  for (int dt = 0; dt < 4; ++dt)
    #pragma unroll
    for (int r = 0; r < 8; ++r) {
      float v = O[dt][r];
      v = v > 0.f ? v : 0.f;
      netR[(rowbase + r + hi * 8) * 512 + wave * 64 + dt * 16 + hl] =
          (_Float16)v;
    }
}

// ---------------------------------------------------------------------------
// Kernel 5: projection GEMM: out[b,c,n] = relu(net)[bn,512] @ w_proj^T + b_proj
// One WG = 16 rows; 8 waves cover the 128 output channels (16 each).
// ---------------------------------------------------------------------------
__global__ __launch_bounds__(256) void proj_kernel(
    const _Float16* __restrict__ netR, const _Float16* __restrict__ wpf,
    const float* __restrict__ bp, float* __restrict__ out)
{
  const int tid = threadIdx.x;
  const int wave = tid >> 5, lane = tid & 31;
  const int hl = lane & 15, hi = lane >> 4;
  const int row0 = blockIdx.x * 16;          // rows over b*N (N % 16 == 0)
  const int b = row0 / NPIX, n0 = row0 % NPIX;
  const int c0 = wave * 16;

  v8f acc = {};
  for (int kc = 0; kc < 16; ++kc) {
    int k0 = kc * 32;
    const _Float16* arow = netR + (size_t)(row0 + hl) * 512 + k0 + hi * 8;
    v8h a0 = *(const v8h*)arow;
    v8h a1 = *(const v8h*)(arow + 16);
    v16h av;
    #pragma unroll
    for (int i = 0; i < 8; ++i) { av[i] = a0[i]; av[8 + i] = a1[i]; }
    const _Float16* brow = wpf + (size_t)(c0 + hl) * 512 + k0 + hi * 8;
    v8h b0 = *(const v8h*)brow;
    v8h b1 = *(const v8h*)(brow + 16);
    v16h bv;
    #pragma unroll
    for (int i = 0; i < 8; ++i) { bv[i] = b0[i]; bv[8 + i] = b1[i]; }
    acc = wmma16(av, bv, acc);
  }
  const int c = c0 + hl;
  const float bb = bp[c];
  #pragma unroll
  for (int r = 0; r < 8; ++r)
    out[((size_t)b * 128 + c) * NPIX + n0 + r + hi * 8] = acc[r] + bb;
}

// ---------------------------------------------------------------------------
extern "C" void kernel_launch(void* const* d_in, const int* in_sizes, int n_in,
                              void* d_out, int out_size, void* d_ws,
                              size_t ws_size, hipStream_t stream)
{
  (void)in_sizes; (void)n_in; (void)out_size; (void)ws_size;
  const float* x     = (const float*)d_in[0];
  const float* wq    = (const float*)d_in[1];
  const float* bq    = (const float*)d_in[2];
  const float* wk    = (const float*)d_in[3];
  const float* bk    = (const float*)d_in[4];
  const float* wv    = (const float*)d_in[5];
  const float* bv    = (const float*)d_in[6];
  const float* w_vl  = (const float*)d_in[7];
  const float* b_vl  = (const float*)d_in[8];
  const float* w_th1 = (const float*)d_in[9];
  const float* b_th1 = (const float*)d_in[10];
  const float* w_th2 = (const float*)d_in[11];
  const float* b_th2 = (const float*)d_in[12];
  const float* w_pr  = (const float*)d_in[13];
  const float* b_pr  = (const float*)d_in[14];

  char* ws = (char*)d_ws;
  _Float16* qf   = (_Float16*)(ws + 0);          // 1,605,632 B
  _Float16* kf   = (_Float16*)(ws + 1605632);    // 1,605,632 B
  float*    vt   = (float*)   (ws + 3211264);    // 12,845,056 B
  _Float16* vf   = (_Float16*)(ws + 16056320);   // 6,422,528 B
  _Float16* wpf  = (_Float16*)(ws + 22478848);   // 131,072 B
  _Float16* netR = (_Float16*)(ws + 22609920);   // 6,422,528 B

  qkv_kernel<<<196, 256, 0, stream>>>(x, wq, bq, wk, bk, wv, bv, qf, kf, vt);
  dw_kernel<<<(2 * 512 * NPIX + 255) / 256, 256, 0, stream>>>(vt, w_vl, b_vl, vf);
  cvt_kernel<<<256, 256, 0, stream>>>(w_pr, wpf);
  attn_kernel<<<2 * (NPIX / 16), 256, 0, stream>>>(qf, kf, vf, w_th1, b_th1,
                                                   w_th2, b_th2, netR);
  proj_kernel<<<(2 * NPIX) / 16, 256, 0, stream>>>(netR, wpf, b_pr,
                                                   (float*)d_out);
}
